// VanillaFlashDiff1_18399639896723
// MI455X (gfx1250) — compile-verified, matrix operand order checked by reading, not run on hardware
//
#include <hip/hip_runtime.h>
#include <hip/hip_bf16.h>
#include <cstdint>

#define Bsz    2
#define Sseq   2048
#define Dmodel 2048
#define Hn     16
#define DHd    128
#define Mrows  (Bsz * Sseq) // 4096

typedef __attribute__((ext_vector_type(16))) __bf16 v16bf;
typedef __attribute__((ext_vector_type(8)))  float  v8f;
typedef __attribute__((ext_vector_type(4)))  int    v4i;

union FragAB { v16bf v; uint32_t u[8]; uint16_t s[16]; };
union FragC  { v8f   v; float    f[8]; };

__device__ __forceinline__ uint16_t f2bf(float f) {
    uint32_t u = __float_as_uint(f);
    u += 0x7fffu + ((u >> 16) & 1u);           // round-to-nearest-even
    return (uint16_t)(u >> 16);
}

// ISA 7.12.2: 16-bit A-matrix 16x32 K index for register-pair i, lane-half h
__device__ __forceinline__ int kmap(int i, int half) {
    return 2 * i + ((i >= 4) ? 8 : 0) + 8 * half;
}

// ---- gfx1250 async copy to LDS (ASYNCcnt path), with safe fallback --------
#if defined(__has_builtin)
#if __has_builtin(__builtin_amdgcn_global_load_async_to_lds_b128)
#define HAS_ASYNC_LDS 1
#endif
#endif
#ifndef HAS_ASYNC_LDS
#define HAS_ASYNC_LDS 0
#endif

__device__ __forceinline__ void async_copy_b128(const uint16_t* g, uint16_t* l) {
#if HAS_ASYNC_LDS
    __builtin_amdgcn_global_load_async_to_lds_b128((v4i*)g, (v4i*)l, 0, 0);
#else
    *(uint4*)l = *(const uint4*)g;
#endif
}

template<int N> __device__ __forceinline__ void wait_async() {
#if defined(__has_builtin) && __has_builtin(__builtin_amdgcn_s_wait_asynccnt)
    __builtin_amdgcn_s_wait_asynccnt(N);
#else
    if constexpr (N == 0) asm volatile("s_wait_asynccnt 0x0" ::: "memory");
    else                  asm volatile("s_wait_asynccnt 0x3" ::: "memory");
#endif
}

// ---------------------------------------------------------------------------
// Prologue casts (bandwidth-trivial: ~100 MB @ 23.3 TB/s)
// ---------------------------------------------------------------------------
__global__ __launch_bounds__(256)
void cast_f32_bf16(const float* __restrict__ in, uint16_t* __restrict__ out) {
    int i = (blockIdx.x * 256 + threadIdx.x) * 4;
    float4 v = *(const float4*)(in + i);
    ushort4 o;
    o.x = f2bf(v.x); o.y = f2bf(v.y); o.z = f2bf(v.z); o.w = f2bf(v.w);
    *(ushort4*)(out + i) = o;
}

// W[k][n] fp32 -> Wt[n][k] bf16 (so GEMM B-tiles are contiguous rows)
__global__ __launch_bounds__(256)
void transpose_cast_w(const float* __restrict__ W, uint16_t* __restrict__ Wt) {
    __shared__ float tile[32][33];
    const int bn = blockIdx.x * 32;     // n block
    const int bk = blockIdx.y * 32;     // k block
    const int tx = threadIdx.x & 31, ty = threadIdx.x >> 5;   // ty in 0..7
    #pragma unroll
    for (int j = 0; j < 32; j += 8)
        tile[ty + j][tx] = W[(size_t)(bk + ty + j) * Dmodel + bn + tx];
    __syncthreads();
    #pragma unroll
    for (int j = 0; j < 32; j += 8)
        Wt[(size_t)(bn + ty + j) * Dmodel + bk + tx] = f2bf(tile[tx][ty + j]);
}

// ---------------------------------------------------------------------------
// bf16 WMMA GEMM, double-buffered async-LDS pipeline.
// C[4096x2048] = A[4096x2048](bf16, row-major) * Wt[2048x2048](bf16, [n][k])
// MODE 1: fused interleaved RoPE, bf16 scatter to [B,H,S,DH]   (Q, K)
// MODE 2: bf16 scatter to [B,H,DH,S] (V transposed for PV WMMA)
// MODE 3: fp32 row-major [M,N]                                  (out)
// ---------------------------------------------------------------------------
template<int MODE>
__global__ __launch_bounds__(256)
void gemm_bf16_wmma(const uint16_t* __restrict__ A, const uint16_t* __restrict__ Wt,
                    void* __restrict__ Out)
{
    __shared__ uint16_t As[2][128][32];   // [m][k] bf16, double buffered
    __shared__ uint16_t Bs[2][64][32];    // [n][k] bf16, double buffered

    const int tid  = threadIdx.x;
    const int wid  = tid >> 5, lane = tid & 31;
    const int half = lane >> 4, lcol = lane & 15;
    const int blockM = blockIdx.y * 128;
    const int blockN = blockIdx.x * 64;
    const int waveM  = (wid & 3) * 32;
    const int waveN  = (wid >> 2) * 32;

    // per-tile async issue: 3 B128 instructions per wave (2 for A, 1 for B)
    auto issue_tile = [&](int kt, int buf) {
        #pragma unroll
        for (int j = 0; j < 2; ++j) {
            int c   = tid * 2 + j;            // 0..511 16-byte chunks of A tile
            int row = c >> 2, kc = (c & 3) * 8;
            async_copy_b128(A + (size_t)(blockM + row) * Dmodel + kt * 32 + kc,
                            &As[buf][row][kc]);
        }
        int n = tid >> 2, kc = (tid & 3) * 8; // 256 chunks of B tile
        async_copy_b128(Wt + (size_t)(blockN + n) * Dmodel + kt * 32 + kc,
                        &Bs[buf][n][kc]);
    };

    FragC acc[2][2];
    #pragma unroll
    for (int m = 0; m < 2; ++m)
        #pragma unroll
        for (int n = 0; n < 2; ++n)
            #pragma unroll
            for (int r = 0; r < 8; ++r) acc[m][n].f[r] = 0.0f;

    const int NT = Dmodel / 32;
    issue_tile(0, 0);
    for (int kt = 0; kt < NT; ++kt) {
        const int buf = kt & 1;
        if (kt + 1 < NT) {                 // prefetch next tile, then wait for
            issue_tile(kt + 1, buf ^ 1);   // current (async done in-order)
            wait_async<3>();
        } else {
            wait_async<0>();
        }
        __syncthreads();

        FragAB a[2], b[2];
        #pragma unroll
        for (int m = 0; m < 2; ++m) {
            int row = waveM + m * 16 + lcol;
            #pragma unroll
            for (int i = 0; i < 8; ++i)
                a[m].u[i] = *(const uint32_t*)&As[buf][row][kmap(i, half)];
        }
        #pragma unroll
        for (int n = 0; n < 2; ++n) {
            int col = waveN + n * 16 + lcol;
            #pragma unroll
            for (int i = 0; i < 8; ++i)
                b[n].u[i] = *(const uint32_t*)&Bs[buf][col][kmap(i, half)];
        }
        #pragma unroll
        for (int m = 0; m < 2; ++m)
            #pragma unroll
            for (int n = 0; n < 2; ++n)
                acc[m][n].v = __builtin_amdgcn_wmma_f32_16x16x32_bf16(
                    false, a[m].v, false, b[n].v, (short)0, acc[m][n].v,
                    false, false);
        __syncthreads();
    }

    // epilogue
    #pragma unroll
    for (int m = 0; m < 2; ++m)
        #pragma unroll
        for (int n = 0; n < 2; ++n)
            #pragma unroll
            for (int r = 0; r < 8; ++r) {
                int row = blockM + waveM + m * 16 + r + 8 * half;
                int col = blockN + waveN + n * 16 + lcol;
                float val = acc[m][n].f[r];
                if (MODE == 3) {
                    ((float*)Out)[(size_t)row * Dmodel + col] = val;
                } else if (MODE == 2) {
                    int b_ = row >> 11, s_ = row & (Sseq - 1);
                    int h_ = col >> 7,  d_ = col & (DHd - 1);
                    ((uint16_t*)Out)[((size_t)(b_ * Hn + h_) * DHd + d_) * Sseq + s_]
                        = f2bf(val);
                } else { // MODE 1: interleaved RoPE via lane-pair exchange
                    float partner = __shfl_xor(val, 1, 32);
                    int b_ = row >> 11, s_ = row & (Sseq - 1);
                    int h_ = col >> 7,  d_ = col & (DHd - 1);
                    int p  = d_ >> 1;
                    // angle = s * 10000^(-2p/128) = s * exp(-p*ln(1e4)/64)
                    float ang = (float)s_ * __expf(-0.14391156831f * (float)p);
                    float sn, c;
                    __sincosf(ang, &sn, &c);
                    float res = (d_ & 1) ? (partner * sn + val * c)
                                         : (val * c - partner * sn);
                    ((uint16_t*)Out)[((size_t)(b_ * Hn + h_) * Sseq + s_) * DHd + d_]
                        = f2bf(res);
                }
            }
}

// ---------------------------------------------------------------------------
// Flash attention (causal), bf16 WMMA, online softmax.
// One wave = 16 query rows; block = 8 waves = 128 rows; grid (S/128, B*H).
// Q/K layout [B,H,S,DH] bf16; V transposed [B,H,DH,S] bf16.
// ---------------------------------------------------------------------------
__global__ __launch_bounds__(256)
void flash_attn_wmma(const uint16_t* __restrict__ Qb,
                     const uint16_t* __restrict__ Kb,
                     const uint16_t* __restrict__ Vt,
                     uint16_t* __restrict__ attn)
{
    __shared__ uint16_t Pld[8][16][32];   // per-wave P re-stripe buffer

    const int tid  = threadIdx.x, wid = tid >> 5, lane = tid & 31;
    const int half = lane >> 4, lcol = lane & 15;
    const int bh = blockIdx.y, b_ = bh >> 4, h_ = bh & (Hn - 1);
    const int qbase = blockIdx.x * 128 + wid * 16;
    const size_t qkBase = (size_t)bh * Sseq * DHd;
    const size_t vBase  = (size_t)bh * DHd * Sseq;
    const float scale = 0.08838834764831845f; // 1/sqrt(128)

    FragAB aQ[4];
    #pragma unroll
    for (int c = 0; c < 4; ++c) {
        const uint16_t* src = Qb + qkBase + (size_t)(qbase + lcol) * DHd + c * 32;
        #pragma unroll
        for (int i = 0; i < 8; ++i)
            aQ[c].u[i] = *(const uint32_t*)&src[kmap(i, half)];
    }

    FragC acc[8];
    float m_i[8], l_i[8];
    #pragma unroll
    for (int r = 0; r < 8; ++r) { m_i[r] = -3.0e38f; l_i[r] = 0.0f; }
    #pragma unroll
    for (int j = 0; j < 8; ++j)
        #pragma unroll
        for (int r = 0; r < 8; ++r) acc[j].f[r] = 0.0f;

    const int qmax = qbase + 15;
    for (int kb = 0; kb <= qmax; kb += 32) {
        FragC sc[2];
        #pragma unroll
        for (int t = 0; t < 2; ++t)
            #pragma unroll
            for (int r = 0; r < 8; ++r) sc[t].f[r] = 0.0f;

        #pragma unroll
        for (int c = 0; c < 4; ++c) {
            FragAB bK0, bK1;
            const uint16_t* s0 = Kb + qkBase + (size_t)(kb + lcol) * DHd + c * 32;
            const uint16_t* s1 = s0 + 16 * DHd;
            #pragma unroll
            for (int i = 0; i < 8; ++i) {
                bK0.u[i] = *(const uint32_t*)&s0[kmap(i, half)];
                bK1.u[i] = *(const uint32_t*)&s1[kmap(i, half)];
            }
            sc[0].v = __builtin_amdgcn_wmma_f32_16x16x32_bf16(
                false, aQ[c].v, false, bK0.v, (short)0, sc[0].v, false, false);
            sc[1].v = __builtin_amdgcn_wmma_f32_16x16x32_bf16(
                false, aQ[c].v, false, bK1.v, (short)0, sc[1].v, false, false);
        }

        #pragma unroll
        for (int r = 0; r < 8; ++r) {
            int row = qbase + r + 8 * half;
            float s0 = sc[0].f[r] * scale;
            float s1 = sc[1].f[r] * scale;
            if (kb + lcol      > row) s0 = -3.0e38f;   // causal mask
            if (kb + 16 + lcol > row) s1 = -3.0e38f;
            float mx = fmaxf(s0, s1);
            #pragma unroll
            for (int off = 1; off < 16; off <<= 1)
                mx = fmaxf(mx, __shfl_xor(mx, off, 32));
            float m_new = fmaxf(m_i[r], mx);
            float alpha = __expf(m_i[r] - m_new);
            float p0 = __expf(s0 - m_new);
            float p1 = __expf(s1 - m_new);
            float rs = p0 + p1;
            #pragma unroll
            for (int off = 1; off < 16; off <<= 1)
                rs += __shfl_xor(rs, off, 32);
            l_i[r] = l_i[r] * alpha + rs;
            m_i[r] = m_new;
            #pragma unroll
            for (int j = 0; j < 8; ++j) acc[j].f[r] *= alpha;
            Pld[wid][r + 8 * half][lcol]      = f2bf(p0);
            Pld[wid][r + 8 * half][lcol + 16] = f2bf(p1);
        }
        __builtin_amdgcn_wave_barrier();   // keep DS store->load ordering

        FragAB aP;
        #pragma unroll
        for (int i = 0; i < 8; ++i)
            aP.u[i] = *(const uint32_t*)&Pld[wid][lcol][kmap(i, half)];

        #pragma unroll
        for (int j = 0; j < 8; ++j) {
            FragAB bV;
            const uint16_t* vs = Vt + vBase + (size_t)(j * 16 + lcol) * Sseq + kb;
            #pragma unroll
            for (int i = 0; i < 8; ++i)
                bV.u[i] = *(const uint32_t*)&vs[kmap(i, half)];
            acc[j].v = __builtin_amdgcn_wmma_f32_16x16x32_bf16(
                false, aP.v, false, bV.v, (short)0, acc[j].v, false, false);
        }
    }

    #pragma unroll
    for (int r = 0; r < 8; ++r) {
        float inv = 1.0f / l_i[r];
        int s_ = qbase + r + 8 * half;
        #pragma unroll
        for (int j = 0; j < 8; ++j)
            attn[(size_t)(b_ * Sseq + s_) * Dmodel + h_ * DHd + j * 16 + lcol]
                = f2bf(acc[j].f[r] * inv);
    }
}

// ---------------------------------------------------------------------------
extern "C" void kernel_launch(void* const* d_in, const int* in_sizes, int n_in,
                              void* d_out, int out_size, void* d_ws, size_t ws_size,
                              hipStream_t stream)
{
    (void)in_sizes; (void)n_in; (void)out_size; (void)ws_size;
    const float* x  = (const float*)d_in[0];
    const float* wq = (const float*)d_in[1];
    const float* wk = (const float*)d_in[2];
    const float* wv = (const float*)d_in[3];
    const float* wo = (const float*)d_in[4];

    char* ws = (char*)d_ws;
    const size_t tsz = (size_t)Bsz * Hn * Sseq * DHd * sizeof(uint16_t); // 16 MB
    const size_t wsz = (size_t)Dmodel * Dmodel * sizeof(uint16_t);       //  8 MB
    uint16_t* Qb   = (uint16_t*)(ws);
    uint16_t* Kb   = (uint16_t*)(ws + tsz);
    uint16_t* Vt   = (uint16_t*)(ws + 2 * tsz);
    uint16_t* attn = (uint16_t*)(ws + 3 * tsz);
    uint16_t* xb   = (uint16_t*)(ws + 4 * tsz);
    uint16_t* wqT  = (uint16_t*)(ws + 5 * tsz);
    uint16_t* wkT  = (uint16_t*)(ws + 5 * tsz + wsz);
    uint16_t* wvT  = (uint16_t*)(ws + 5 * tsz + 2 * wsz);
    uint16_t* woT  = (uint16_t*)(ws + 5 * tsz + 3 * wsz);

    dim3 blk(256);

    // prologue casts
    cast_f32_bf16<<<(Mrows * Dmodel) / (256 * 4), blk, 0, stream>>>(x, xb);
    dim3 tGrid(Dmodel / 32, Dmodel / 32);
    transpose_cast_w<<<tGrid, blk, 0, stream>>>(wq, wqT);
    transpose_cast_w<<<tGrid, blk, 0, stream>>>(wk, wkT);
    transpose_cast_w<<<tGrid, blk, 0, stream>>>(wv, wvT);
    transpose_cast_w<<<tGrid, blk, 0, stream>>>(wo, woT);

    dim3 gGrid(Dmodel / 64, Mrows / 128);               // 32 x 32 blocks
    gemm_bf16_wmma<1><<<gGrid, blk, 0, stream>>>(xb, wqT, Qb);   // Q + RoPE
    gemm_bf16_wmma<1><<<gGrid, blk, 0, stream>>>(xb, wkT, Kb);   // K + RoPE
    gemm_bf16_wmma<2><<<gGrid, blk, 0, stream>>>(xb, wvT, Vt);   // V^T
    flash_attn_wmma<<<dim3(Sseq / 128, Bsz * Hn), blk, 0, stream>>>(Qb, Kb, Vt, attn);
    gemm_bf16_wmma<3><<<gGrid, blk, 0, stream>>>(attn, woT, (float*)d_out);
}
